// RelationCos_11364483465329
// MI455X (gfx1250) — matrix operand order repaired and checked by reading, not compiled, not run on hardware
//
#include <hip/hip_runtime.h>
#include <hip/hip_bf16.h>
#include <stdint.h>

// ---------------- CDNA5 WMMA types ----------------
typedef __attribute__((ext_vector_type(16))) __bf16 v16bf;
typedef __attribute__((ext_vector_type(8)))  __bf16 v8bf;
typedef __attribute__((ext_vector_type(8)))  float  v8f;

#define N_PTS  8192
#define BATCH  16
#define S_SAMP 32
#define KNN_K  12
#define D_OUT  1024
#define NGROUP (BATCH * S_SAMP)   // 512 output rows per branch

// ---------------------------------------------------------------------------
// 1) Furthest point sampling: one block per batch, dist[] in LDS.
//    Matches jax scan semantics: start index 0, 31 argmax steps, first-index ties.
// ---------------------------------------------------------------------------
__global__ void fps_kernel(const float* __restrict__ xyz, float* __restrict__ new_xyz) {
    __shared__ float dist[N_PTS];
    __shared__ float rv[256];
    __shared__ int   ri[256];
    __shared__ float lastp[3];
    const int b   = blockIdx.x;
    const int tid = threadIdx.x;
    const float* P = xyz + (size_t)b * N_PTS * 3;

    for (int i = tid; i < N_PTS; i += 256) dist[i] = 1e10f;
    if (tid == 0) {
        lastp[0] = P[0]; lastp[1] = P[1]; lastp[2] = P[2];
        float* o = new_xyz + (size_t)(b * S_SAMP) * 3;
        o[0] = P[0]; o[1] = P[1]; o[2] = P[2];
    }
    __syncthreads();

    for (int s = 1; s < S_SAMP; ++s) {
        const float lx = lastp[0], ly = lastp[1], lz = lastp[2];
        float bv = -1.0f; int bi = 0x7fffffff;
        for (int i = tid; i < N_PTS; i += 256) {
            const float dx = P[i*3+0]-lx, dy = P[i*3+1]-ly, dz = P[i*3+2]-lz;
            const float d  = dx*dx + dy*dy + dz*dz;
            const float nd = fminf(dist[i], d);
            dist[i] = nd;
            if (nd > bv || (nd == bv && i < bi)) { bv = nd; bi = i; }
        }
        rv[tid] = bv; ri[tid] = bi;
        __syncthreads();
        for (int off = 128; off > 0; off >>= 1) {
            if (tid < off) {
                const float ov = rv[tid+off]; const int oi = ri[tid+off];
                if (ov > rv[tid] || (ov == rv[tid] && oi < ri[tid])) { rv[tid] = ov; ri[tid] = oi; }
            }
            __syncthreads();
        }
        if (tid == 0) {
            const int li = ri[0];
            lastp[0] = P[li*3+0]; lastp[1] = P[li*3+1]; lastp[2] = P[li*3+2];
            float* o = new_xyz + (size_t)(b * S_SAMP + s) * 3;
            o[0] = lastp[0]; o[1] = lastp[1]; o[2] = lastp[2];
        }
        __syncthreads();
    }
}

// ---------------------------------------------------------------------------
// 2) KNN: one block per (b, m); all 8192 sq-dists in LDS, 12 argmin rounds.
// ---------------------------------------------------------------------------
__global__ void knn_kernel(const float* __restrict__ xyz, const float* __restrict__ new_xyz,
                           int* __restrict__ idx) {
    __shared__ float sd[N_PTS];
    __shared__ float rv[256];
    __shared__ int   ri[256];
    const int bm  = blockIdx.x;          // b*32 + m
    const int b   = bm >> 5;
    const int tid = threadIdx.x;
    const float* P = xyz + (size_t)b * N_PTS * 3;
    const float qx = new_xyz[bm*3+0], qy = new_xyz[bm*3+1], qz = new_xyz[bm*3+2];

    for (int i = tid; i < N_PTS; i += 256) {
        const float dx = P[i*3+0]-qx, dy = P[i*3+1]-qy, dz = P[i*3+2]-qz;
        sd[i] = dx*dx + dy*dy + dz*dz;
    }
    __syncthreads();

    for (int j = 0; j < KNN_K; ++j) {
        float bv = 3.0e38f; int bi = 0x7fffffff;
        for (int i = tid; i < N_PTS; i += 256) {
            const float v = sd[i];
            if (v < bv || (v == bv && i < bi)) { bv = v; bi = i; }
        }
        rv[tid] = bv; ri[tid] = bi;
        __syncthreads();
        for (int off = 128; off > 0; off >>= 1) {
            if (tid < off) {
                const float ov = rv[tid+off]; const int oi = ri[tid+off];
                if (ov < rv[tid] || (ov == rv[tid] && oi < ri[tid])) { rv[tid] = ov; ri[tid] = oi; }
            }
            __syncthreads();
        }
        if (tid == 0) {
            idx[bm * KNN_K + j] = ri[0];
            sd[ri[0]] = 3.0e38f;   // exclude
        }
        __syncthreads();
    }
}

// ---------------------------------------------------------------------------
// 3a) fp32 -> bf16 weight conversion (W kept [o, d] row-major).
// ---------------------------------------------------------------------------
__global__ void cvt_bf16_kernel(const float* __restrict__ src, __bf16* __restrict__ dst, int count) {
    const int i = blockIdx.x * 256 + threadIdx.x;
    if (i < count) dst[i] = (__bf16)src[i];
}

// 3b) Gather features into Ag[k][g][d] (bf16): one block per (k,g) row.
__global__ void gather_bf16_kernel(const float* __restrict__ feat, const int* __restrict__ idx,
                                   __bf16* __restrict__ dst, int d) {
    const int row = blockIdx.x;          // 0 .. 12*512-1, row = k*512 + g
    const int k   = row / NGROUP;
    const int g   = row - k * NGROUP;    // b*32 + m
    const int b   = g >> 5;
    const int src = idx[g * KNN_K + k];
    const float* srow = feat + ((size_t)b * N_PTS + (size_t)src) * d;
    __bf16* drow = dst + (size_t)row * d;
    for (int c = threadIdx.x; c < d; c += 256) drow[c] = (__bf16)srow[c];
}

// ---------------------------------------------------------------------------
// 4) Fused WMMA GEMM + BN(eval) + ReLU + max over k.
//    One wave -> one 16(g) x 16(o) output tile. 2048 waves = 256 blocks x 8 waves.
//    A fragment layout (16-bit A 16x32): lane M = lane&15,
//      16 contiguous K at half*16 (matches B).
//    B fragment (32x16): lane N = lane&15, 16 contiguous K at half*16.
//    C/D: VGPR r, lane -> M = r + 8*half, N = lane&15.
// ---------------------------------------------------------------------------
__global__ void gemm_bn_relu_max_kernel(
    const __bf16* __restrict__ Ag,     // [12, 512, d]
    const __bf16* __restrict__ Wb,     // [1024, d]
    const float* __restrict__ bias, const float* __restrict__ gamma,
    const float* __restrict__ beta, const float* __restrict__ mean,
    const float* __restrict__ var,
    float* __restrict__ out,           // [512, 1024]
    int d)
{
    const int lane = threadIdx.x & 31;
    const int wave = blockIdx.x * (blockDim.x >> 5) + (threadIdx.x >> 5);
    const int gt   = wave >> 6;        // 0..31 : which 16-row group tile
    const int ot   = wave & 63;        // 0..63 : which 16-channel tile
    const int n    = lane & 15;
    const int half = lane >> 4;
    const int o    = ot * 16 + n;

    // Folded BN: y = dot*sc + sh  (then ReLU).
    const float sc = gamma[o] * rsqrtf(var[o] + 1e-5f);
    const float sh = (bias[o] - mean[o]) * sc + beta[o];

    const __bf16* Bbase = Wb + (size_t)o * d + half * 16;
    const __bf16* Abase = Ag + (size_t)(gt * 16 + n) * d + half * 16;
    const size_t  Akstride = (size_t)NGROUP * d;   // advance one k plane

    v8f vmax = {};
    for (int k = 0; k < KNN_K; ++k) {
        const __bf16* Arow = Abase + (size_t)k * Akstride;
        v8f acc = {};
        for (int d0 = 0; d0 < d; d0 += 32) {
            const v16bf a = *(const v16bf*)(Arow + d0);
            const v16bf bmat = *(const v16bf*)(Bbase + d0);
            acc = __builtin_amdgcn_wmma_f32_16x16x32_bf16(
                false, a, false, bmat, (short)0, acc, false, false);
        }
#pragma unroll
        for (int r = 0; r < 8; ++r) {
            float y = acc[r] * sc + sh;
            y = fmaxf(y, 0.0f);
            vmax[r] = fmaxf(vmax[r], y);
        }
    }
#pragma unroll
    for (int r = 0; r < 8; ++r) {
        const int g = gt * 16 + r + 8 * half;
        out[(size_t)g * D_OUT + o] = vmax[r];
    }
}

// ---------------------------------------------------------------------------
// Host launcher
// ---------------------------------------------------------------------------
extern "C" void kernel_launch(void* const* d_in, const int* in_sizes, int n_in,
                              void* d_out, int out_size, void* d_ws, size_t ws_size,
                              hipStream_t stream) {
    (void)in_sizes; (void)n_in; (void)out_size; (void)ws_size;
    const float* feature_s = (const float*)d_in[0];   // [16,8192,256]
    const float* xyz_s     = (const float*)d_in[1];   // [16,8192,3]
    const float* feature_t = (const float*)d_in[2];   // [16,8192,1024]
    const float* xyz_t     = (const float*)d_in[3];   // [16,8192,3]
    const float* Ws        = (const float*)d_in[4];   // [1024,256]
    const float* bias_s    = (const float*)d_in[5];
    const float* gamma_s   = (const float*)d_in[6];
    const float* beta_s    = (const float*)d_in[7];
    const float* mean_s    = (const float*)d_in[8];
    const float* var_s     = (const float*)d_in[9];
    const float* Wt        = (const float*)d_in[10];  // [1024,1024]
    const float* bias_t    = (const float*)d_in[11];
    const float* gamma_t   = (const float*)d_in[12];
    const float* beta_t    = (const float*)d_in[13];
    const float* mean_t    = (const float*)d_in[14];
    const float* var_t     = (const float*)d_in[15];

    // Workspace carve-out (256B aligned slices)
    uintptr_t cur = (uintptr_t)d_ws;
    auto take = [&cur](size_t bytes) -> void* {
        void* p = (void*)cur;
        cur += (bytes + 255) & ~(size_t)255;
        return p;
    };
    float*  new_xyz = (float*)take((size_t)BATCH * S_SAMP * 3 * sizeof(float));
    int*    idx_t   = (int*)  take((size_t)BATCH * S_SAMP * KNN_K * sizeof(int));
    int*    idx_s   = (int*)  take((size_t)BATCH * S_SAMP * KNN_K * sizeof(int));
    __bf16* Wb_s    = (__bf16*)take((size_t)D_OUT * 256  * sizeof(__bf16));
    __bf16* Wb_t    = (__bf16*)take((size_t)D_OUT * 1024 * sizeof(__bf16));
    __bf16* Ag_s    = (__bf16*)take((size_t)KNN_K * NGROUP * 256  * sizeof(__bf16));
    __bf16* Ag_t    = (__bf16*)take((size_t)KNN_K * NGROUP * 1024 * sizeof(__bf16));

    float* out_s = (float*)d_out;                       // [512,1024]
    float* out_t = out_s + (size_t)NGROUP * D_OUT;      // [512,1024]

    // 1) FPS on xyz_t
    fps_kernel<<<BATCH, 256, 0, stream>>>(xyz_t, new_xyz);
    // 2) KNN against both clouds
    knn_kernel<<<BATCH * S_SAMP, 256, 0, stream>>>(xyz_t, new_xyz, idx_t);
    knn_kernel<<<BATCH * S_SAMP, 256, 0, stream>>>(xyz_s, new_xyz, idx_s);
    // 3) Weight conversion + feature gather (bf16)
    cvt_bf16_kernel<<<(D_OUT * 256  + 255) / 256, 256, 0, stream>>>(Ws, Wb_s, D_OUT * 256);
    cvt_bf16_kernel<<<(D_OUT * 1024 + 255) / 256, 256, 0, stream>>>(Wt, Wb_t, D_OUT * 1024);
    gather_bf16_kernel<<<KNN_K * NGROUP, 256, 0, stream>>>(feature_s, idx_s, Ag_s, 256);
    gather_bf16_kernel<<<KNN_K * NGROUP, 256, 0, stream>>>(feature_t, idx_t, Ag_t, 1024);
    // 4) Fused WMMA GEMM + BN + ReLU + max-over-k   (2048 waves each)
    gemm_bn_relu_max_kernel<<<256, 256, 0, stream>>>(
        Ag_s, Wb_s, bias_s, gamma_s, beta_s, mean_s, var_s, out_s, 256);
    gemm_bn_relu_max_kernel<<<256, 256, 0, stream>>>(
        Ag_t, Wb_t, bias_t, gamma_t, beta_t, mean_t, var_t, out_t, 1024);
}
